// RUMLayer_52166672778203
// MI455X (gfx1250) — matrix operand order repaired and checked by reading, not compiled, not run on hardware
//
#include <hip/hip_runtime.h>
#include <hip/hip_bf16.h>
#include <math.h>

typedef __attribute__((ext_vector_type(16))) _Float16 v16h;
typedef __attribute__((ext_vector_type(8)))  _Float16 v8h;
typedef __attribute__((ext_vector_type(8)))  float    v8f;

#define N_NODES_  8000
#define N_EDGES_  64000
#define NSAMP_    4
#define LSEQ_     8
#define INF_      64
#define OUTF_     64
#define GD_       192          // 3 * OUTF
#define DMAIN_    193
#define TSEQ_     16           // sequences per wave-tile
#define LDH_      72           // padded LDS row (halves)
#define NB_       (NSAMP_ * N_NODES_)   // 32000
#define NTILE_    (NB_ / TSEQ_)         // 2000

// ---------------- workspace layout (bytes) ----------------
#define OFF_XGE   ((size_t)0)
#define SZ_XGE    ((size_t)N_EDGES_ * GD_ * 4)          // 49,152,000
#define OFF_XGH   (OFF_XGE + SZ_XGE)
#define SZ_XGH    ((size_t)N_NODES_ * GD_ * 4)          //  6,144,000
#define OFF_WC    (OFF_XGH + SZ_XGH)
#define SZ_WC     ((size_t)GD_ * 16 * 4)
#define OFF_DMAX  (OFF_WC + SZ_WC)
#define OFF_PACK  (OFF_DMAX + 256)
#define SZ_PACK1  ((size_t)GD_ * OUTF_)                 // halves per Whh-style matrix (12288)
#define OFF_PACKX (OFF_PACK + 5 * SZ_PACK1 * 2)         // after 5 matrices (bytes)
#define SZ_PACKX1 ((size_t)12 * 512)                    // halves per x-matrix (6144)

// ---------------- helpers ----------------
__device__ __forceinline__ v16h a_frag(const _Float16* mat, int kb, int lane) {
    // A 16x32 f16 layout (contiguous K): row r = lane&15; hi = lane>>4;
    // element i : K = kb*32 + 16*hi + i
    int r = lane & 15, hi = lane >> 4;
    const _Float16* p = mat + r * LDH_ + kb * 32 + 16 * hi;
    v8h lo = *(const v8h*)(p);
    v8h hv = *(const v8h*)(p + 8);
    v16h a;
#pragma unroll
    for (int i = 0; i < 8; ++i) { a[i] = lo[i]; a[i + 8] = hv[i]; }
    return a;
}

__device__ __forceinline__ v16h b_frag(const _Float16* pk, int jj, int kb, int lane) {
    // pre-packed: frag (jj,kb): 32 lanes x 16 contiguous halves
    return *(const v16h*)(pk + ((((jj << 1) | kb) << 5) + lane) * 16);
}
__device__ __forceinline__ v16h bx_frag(const _Float16* pk, int jj, int lane) {
    // x-projection matrices: single kb per jj
    return *(const v16h*)(pk + (((jj) << 5) + lane) * 16);
}

__device__ __forceinline__ v8f wmma_acc(v16h a, v16h b, v8f c) {
    return __builtin_amdgcn_wmma_f32_16x16x32_f16(false, a, false, b, (short)0, c, false, false);
}
// zero-C WMMA: zeroinitializer C folds to inline SRC2=0 (no broadcast movs)
__device__ __forceinline__ v8f wmma_z(v16h a, v16h b) {
    v8f z = {};
    return __builtin_amdgcn_wmma_f32_16x16x32_f16(false, a, false, b, (short)0, z, false, false);
}

#define LOG2E_ 1.4426950408889634f
__device__ __forceinline__ float sigm_(float x) {
    return __builtin_amdgcn_rcpf(1.f + __builtin_amdgcn_exp2f(-LOG2E_ * x));
}
__device__ __forceinline__ float tanh_(float x) {
    return 1.f - 2.f * __builtin_amdgcn_rcpf(1.f + __builtin_amdgcn_exp2f((2.f * LOG2E_) * x));
}

// ---------------- phase 0: W_combo = Wih_m @ W_edge, init degmax ----------------
__global__ void k_wcombo(const float* __restrict__ WihM, const float* __restrict__ Wedge,
                         float* __restrict__ wc, int* __restrict__ dmax) {
    int idx = blockIdx.x * blockDim.x + threadIdx.x;
    if (idx == 0) *dmax = 0;
    if (idx >= GD_ * 16) return;
    int g = idx >> 4, f = idx & 15;
    float s = 0.f;
    for (int c = 0; c < DMAIN_; ++c) s += WihM[g * DMAIN_ + c] * Wedge[c * 16 + f];
    wc[idx] = s;
}

// ---------------- phase 0: max over node_degrees[walks] ----------------
__global__ void k_dmax(const int* __restrict__ walks, const int* __restrict__ deg,
                       int* __restrict__ dmax) {
    __shared__ int red[256];
    int idx = blockIdx.x * 256 + threadIdx.x;
    int v = 0;
    if (idx < NB_ * LSEQ_) v = deg[walks[idx]];
    red[threadIdx.x] = v;
    __syncthreads();
    for (int s = 128; s > 0; s >>= 1) {
        if (threadIdx.x < s) red[threadIdx.x] = max(red[threadIdx.x], red[threadIdx.x + s]);
        __syncthreads();
    }
    if (threadIdx.x == 0) atomicMax(dmax, red[0]);
}

// ---------------- phase 0: pack B fragments (f16) for 5 (192x64) matrices ----------------
// order: [0]=Whh_wf [1]=Whh_wb [2]=Whh_m [3]=W1(=Wih_m[:,64:128]) [4]=W2(=Wih_m[:,128:192])
__global__ void k_pack(const float* __restrict__ WhhF, const float* __restrict__ WhhB,
                       const float* __restrict__ WhhM, const float* __restrict__ WihM,
                       _Float16* __restrict__ pk) {
    int idx = blockIdx.x * blockDim.x + threadIdx.x;
    if (idx >= 5 * GD_ * OUTF_) return;
    int mat  = idx / (GD_ * OUTF_);
    int r    = idx % (GD_ * OUTF_);
    int i    = r & 15;
    int lane = (r >> 4) & 31;
    int frag = r >> 9;            // 0..23 : frag = jj*2 + kb
    int kb   = frag & 1;
    int jj   = frag >> 1;
    int g = 16 * jj + (lane & 15);                 // column of B  (gate index)
    int k = kb * 32 + 16 * (lane >> 4) + i;        // row of B     (K index)
    float w;
    if      (mat == 0) w = WhhF[g * OUTF_ + k];
    else if (mat == 1) w = WhhB[g * OUTF_ + k];
    else if (mat == 2) w = WhhM[g * OUTF_ + k];
    else if (mat == 3) w = WihM[g * DMAIN_ + 64 + k];
    else               w = WihM[g * DMAIN_ + 128 + k];
    pk[idx] = (_Float16)w;
}

// ---- phase 0: pack walk-GRU input projection + bias row: B[k][g], k=0,1 = Wih, k=2 = bias ----
__global__ void k_packx(const float* __restrict__ Wih, const float* __restrict__ bih,
                        const float* __restrict__ bhh, _Float16* __restrict__ pk) {
    int idx = blockIdx.x * blockDim.x + threadIdx.x;
    if (idx >= 12 * 512) return;
    int i    = idx & 15;
    int lane = (idx >> 4) & 31;
    int jj   = idx >> 9;                           // 0..11
    int g = 16 * jj + (lane & 15);
    int k = 16 * (lane >> 4) + i;
    float w = 0.f;
    if      (k == 0) w = Wih[2 * g];
    else if (k == 1) w = Wih[2 * g + 1];
    else if (k == 2) w = bih[g] + (g < 2 * OUTF_ ? bhh[g] : 0.f);  // n-gate bhh stays on h-side
    pk[idx] = (_Float16)w;
}

// ---- phase 1: XGE[e][g] = e . W_combo[g] + bih_m[g] + (bhh_m[g] for r/z gates) ----
__global__ void k_xge(const float* __restrict__ e, const float* __restrict__ wc,
                      const float* __restrict__ bihm, const float* __restrict__ bhhm,
                      float* __restrict__ xge) {
    int idx = blockIdx.x * 256 + threadIdx.x;
    if (idx >= N_EDGES_ * GD_) return;
    int eid = idx / GD_, g = idx % GD_;
    float s = bihm[g] + (g < 2 * OUTF_ ? bhhm[g] : 0.f);
    const float* er = e + eid * 16;
    const float* wr = wc + g * 16;
#pragma unroll
    for (int f = 0; f < 16; ++f) s += er[f] * wr[f];
    xge[idx] = s;
}

// ---- phase 1: XGH[n][g] = h[n].Wih_m[g,0:64] + biases + deg_n * Wdeg[g]  (runs after k_dmax) ----
__global__ void k_xgh(const float* __restrict__ h, const float* __restrict__ WihM,
                      const float* __restrict__ bihm, const float* __restrict__ bhhm,
                      const int* __restrict__ ndeg, const int* __restrict__ dmax,
                      float* __restrict__ xgh) {
    int idx = blockIdx.x * 256 + threadIdx.x;
    if (idx >= N_NODES_ * GD_) return;
    int n = idx / GD_, g = idx % GD_;
    int dmv = *dmax;
    float invd = 1.f / (float)(dmv > 0 ? dmv : 1);
    float s = bihm[g] + (g < 2 * OUTF_ ? bhhm[g] : 0.f)
            + (float)ndeg[n] * invd * WihM[g * DMAIN_ + (DMAIN_ - 1)];
    const float* hr = h + n * INF_;
    const float* wr = WihM + g * DMAIN_;
    for (int u = 0; u < INF_; ++u) s += hr[u] * wr[u];
    xgh[idx] = s;
}

// ---------------- phase 2: fused walk GRUs + main GRU, 1 wave = 16 sequences ----------------
__global__ void __launch_bounds__(32) k_rum(
    const int* __restrict__ walks, const int* __restrict__ eids,
    const float* __restrict__ bhh_wf, const float* __restrict__ bhh_wb,
    const float* __restrict__ bhh_m,
    const float* __restrict__ XGH, const float* __restrict__ XGE,
    const _Float16* __restrict__ pWhhF, const _Float16* __restrict__ pWhhB,
    const _Float16* __restrict__ pWhhM,
    const _Float16* __restrict__ pW1,  const _Float16* __restrict__ pW2,
    const _Float16* __restrict__ pWxF, const _Float16* __restrict__ pWxB,
    float* __restrict__ out)
{
    __shared__ __align__(16) _Float16 sH[TSEQ_ * LDH_];
    __shared__ __align__(16) _Float16 sYf[LSEQ_][TSEQ_ * LDH_];
    __shared__ __align__(16) _Float16 sYb[LSEQ_][TSEQ_ * LDH_];
    __shared__ float sHtmp[TSEQ_][OUTF_];
    __shared__ float sUw[LSEQ_][TSEQ_][2];
    __shared__ int   sWalkP[TSEQ_][LSEQ_];
    __shared__ int   sWalkR[LSEQ_][TSEQ_];
    __shared__ int   sEid[LSEQ_ - 1][TSEQ_];
    __shared__ float sBnH[3][OUTF_];       // n-gate h-side bias bhh[128+u] : wf / wb / main

    const int lane = threadIdx.x;
    const int base = blockIdx.x * TSEQ_;
    const int hi   = lane >> 4;
    const int ncol = lane & 15;

    // ---- stage biases / walks / eids ----
    for (int u = lane; u < OUTF_; u += 32) {
        sBnH[0][u] = bhh_wf[2 * OUTF_ + u];
        sBnH[1][u] = bhh_wb[2 * OUTF_ + u];
        sBnH[2][u] = bhh_m[2 * OUTF_ + u];
    }
    for (int idx = lane; idx < TSEQ_ * LSEQ_; idx += 32) {
        int m = idx >> 3, p = idx & 7;
        sWalkP[m][p] = walks[(base + m) * LSEQ_ + p];
    }
    for (int idx = lane; idx < TSEQ_ * (LSEQ_ - 1); idx += 32) {
        int m = idx / 7, t = idx % 7;
        sEid[t][m] = eids[(base + m) * (LSEQ_ - 1) + t];
    }
    __syncthreads();

    // ---- uniq / uw / reversed walk ----
    for (int idx = lane; idx < TSEQ_ * LSEQ_; idx += 32) {
        int m = idx >> 3, p = idx & 7;
        int wid = sWalkP[m][p];
        int q = 0;
        while (sWalkP[m][q] != wid) q++;        // first occurrence (q <= p always)
        float ang = (float)q * (6.283185307179586f / (float)LSEQ_);
        int t = (LSEQ_ - 1) - p;                // reversed time index
        sUw[t][m][0] = sinf(ang);
        sUw[t][m][1] = cosf(ang);
        sWalkR[t][m] = wid;
    }

    float hReg[4][8];

    // ================= walk GRUs: d=0 forward, d=1 backward =================
    for (int d = 0; d < 2; ++d) {
        const _Float16* pWhh = d ? pWhhB : pWhhF;
        const _Float16* pWx  = d ? pWxB  : pWxF;
#pragma unroll
        for (int j = 0; j < 4; ++j)
#pragma unroll
            for (int v = 0; v < 8; ++v) hReg[j][v] = 0.f;
#pragma unroll
        for (int j = 0; j < 4; ++j)
            for (int v = 0; v < 8; ++v) {
                int m = v + 8 * hi, u = 16 * j + ncol;
                sH[m * LDH_ + u] = (_Float16)0.f;
            }
        __syncthreads();

        for (int ti = 0; ti < LSEQ_; ++ti) {
            int t = d ? (LSEQ_ - 1 - ti) : ti;
            // A fragment of [sin, cos, 1, 0...]: only K=0..2 nonzero => hi==0 lanes only
            v16h aU;
#pragma unroll
            for (int i = 0; i < 16; ++i) aU[i] = (_Float16)0.f;
            if (hi == 0) {
                aU[0] = (_Float16)sUw[t][ncol][0];
                aU[1] = (_Float16)sUw[t][ncol][1];
                aU[2] = (_Float16)1.f;
            }
            v16h a0 = a_frag(sH, 0, lane);
            v16h a1 = a_frag(sH, 1, lane);
#pragma unroll
            for (int j = 0; j < 4; ++j) {
                v8f gA[2];    // r,z pre-activations (x + h + biases fully via WMMA)
#pragma unroll
                for (int gi = 0; gi < 2; ++gi) {
                    int jj = j + 4 * gi;
                    v8f c = wmma_z(aU, bx_frag(pWx, jj, lane));        // x part + bias row
                    c = wmma_acc(a0, b_frag(pWhh, jj, 0, lane), c);
                    c = wmma_acc(a1, b_frag(pWhh, jj, 1, lane), c);
                    gA[gi] = c;
                }
                int jn = j + 8;
                v8f xn = wmma_z(aU, bx_frag(pWx, jn, lane));           // xn + bih_n
                v8f hn = wmma_z(a0, b_frag(pWhh, jn, 0, lane));
                hn = wmma_acc(a1, b_frag(pWhh, jn, 1, lane), hn);
                float bn = sBnH[d][16 * j + ncol];
#pragma unroll
                for (int v = 0; v < 8; ++v) {
                    float r = sigm_(gA[0][v]);
                    float z = sigm_(gA[1][v]);
                    float n = tanh_(xn[v] + r * (hn[v] + bn));
                    hReg[j][v] = (1.f - z) * n + z * hReg[j][v];
                }
            }
            __syncthreads();
            _Float16* sY = d ? &sYb[t][0] : &sYf[t][0];
#pragma unroll
            for (int j = 0; j < 4; ++j)
                for (int v = 0; v < 8; ++v) {
                    int m = v + 8 * hi, u = 16 * j + ncol;
                    _Float16 hv = (_Float16)hReg[j][v];
                    sH[m * LDH_ + u] = hv;
                    sY[m * LDH_ + u] = hv;
                }
            __syncthreads();
        }
        if (d == 0) {   // save hf (each lane owns its elements; no barrier needed)
#pragma unroll
            for (int j = 0; j < 4; ++j)
                for (int v = 0; v < 8; ++v) {
                    int m = v + 8 * hi, u = 16 * j + ncol;
                    sHtmp[m][u] = hReg[j][v];
                }
        }
    }

    // ---- h_walk = 0.5*(hf + hb) ----
#pragma unroll
    for (int j = 0; j < 4; ++j)
        for (int v = 0; v < 8; ++v) {
            int m = v + 8 * hi, u = 16 * j + ncol;
            float hw_ = 0.5f * (hReg[j][v] + sHtmp[m][u]);
            hReg[j][v] = hw_;
            sH[m * LDH_ + u] = (_Float16)hw_;
        }
    __syncthreads();

    // ================= main GRU: 15 interleaved steps =================
    for (int s = 0; s < 2 * LSEQ_ - 1; ++s) {
        v16h a0 = a_frag(sH, 0, lane);
        v16h a1 = a_frag(sH, 1, lane);
        const bool even = (s & 1) == 0;
        const int  t    = even ? (s >> 1) : ((s - 1) >> 1);
        v16h yf0, yf1, yb0, yb1;
        if (even) {
            yf0 = a_frag(&sYf[t][0], 0, lane);
            yf1 = a_frag(&sYf[t][0], 1, lane);
            yb0 = a_frag(&sYb[t][0], 0, lane);
            yb1 = a_frag(&sYb[t][0], 1, lane);
        }
#pragma unroll
        for (int j = 0; j < 4; ++j) {
            v8f hA[3], xA[3];
#pragma unroll
            for (int gi = 0; gi < 3; ++gi) {
                int jj = j + 4 * gi;
                int g  = 16 * jj + ncol;
                v8f c = wmma_z(a0, b_frag(pWhhM, jj, 0, lane));
                c = wmma_acc(a1, b_frag(pWhhM, jj, 1, lane), c);
                hA[gi] = c;
                v8f x;
                if (even) {
#pragma unroll
                    for (int v = 0; v < 8; ++v) {
                        int m = v + 8 * hi;
                        x[v] = XGH[sWalkR[t][m] * GD_ + g];   // includes biases + deg term
                    }
                    x = wmma_acc(yf0, b_frag(pW1, jj, 0, lane), x);
                    x = wmma_acc(yf1, b_frag(pW1, jj, 1, lane), x);
                    x = wmma_acc(yb0, b_frag(pW2, jj, 0, lane), x);
                    x = wmma_acc(yb1, b_frag(pW2, jj, 1, lane), x);
                } else {
#pragma unroll
                    for (int v = 0; v < 8; ++v) {
                        int m = v + 8 * hi;
                        x[v] = XGE[sEid[t][m] * GD_ + g];
                    }
                }
                xA[gi] = x;
            }
            float bn = sBnH[2][16 * j + ncol];
#pragma unroll
            for (int v = 0; v < 8; ++v) {
                float r = sigm_(xA[0][v] + hA[0][v]);
                float z = sigm_(xA[1][v] + hA[1][v]);
                float n = tanh_(xA[2][v] + r * (hA[2][v] + bn));
                hReg[j][v] = (1.f - z) * n + z * hReg[j][v];
            }
        }
        __syncthreads();
#pragma unroll
        for (int j = 0; j < 4; ++j)
            for (int v = 0; v < 8; ++v) {
                int m = v + 8 * hi, u = 16 * j + ncol;
                sH[m * LDH_ + u] = (_Float16)hReg[j][v];
            }
        __syncthreads();
    }

    // ---- store h_last ----
#pragma unroll
    for (int j = 0; j < 4; ++j)
        for (int v = 0; v < 8; ++v) {
            int m = v + 8 * hi, u = 16 * j + ncol;
            out[(size_t)(base + m) * OUTF_ + u] = hReg[j][v];
        }
}

// ---------------- launch ----------------
extern "C" void kernel_launch(void* const* d_in, const int* in_sizes, int n_in,
                              void* d_out, int out_size, void* d_ws, size_t ws_size,
                              hipStream_t stream) {
    const float* h      = (const float*)d_in[0];
    const float* e      = (const float*)d_in[2];
    const int*   walks  = (const int*)d_in[3];
    const int*   eids   = (const int*)d_in[4];
    const int*   ndeg   = (const int*)d_in[5];
    const float* Wih_wf = (const float*)d_in[6];
    const float* Whh_wf = (const float*)d_in[7];
    const float* bih_wf = (const float*)d_in[8];
    const float* bhh_wf = (const float*)d_in[9];
    const float* Wih_wb = (const float*)d_in[10];
    const float* Whh_wb = (const float*)d_in[11];
    const float* bih_wb = (const float*)d_in[12];
    const float* bhh_wb = (const float*)d_in[13];
    const float* Wih_m  = (const float*)d_in[14];
    const float* Whh_m  = (const float*)d_in[15];
    const float* bih_m  = (const float*)d_in[16];
    const float* bhh_m  = (const float*)d_in[17];
    const float* Wedge  = (const float*)d_in[18];

    char* ws = (char*)d_ws;
    float*    XGE  = (float*)(ws + OFF_XGE);
    float*    XGH  = (float*)(ws + OFF_XGH);
    float*    WC   = (float*)(ws + OFF_WC);
    int*      DMAX = (int*)(ws + OFF_DMAX);
    _Float16* PK   = (_Float16*)(ws + OFF_PACK);
    _Float16* pWhhF = PK + 0 * SZ_PACK1;
    _Float16* pWhhB = PK + 1 * SZ_PACK1;
    _Float16* pWhhM = PK + 2 * SZ_PACK1;
    _Float16* pW1   = PK + 3 * SZ_PACK1;
    _Float16* pW2   = PK + 4 * SZ_PACK1;
    _Float16* PKX   = (_Float16*)(ws + OFF_PACKX);
    _Float16* pWxF  = PKX + 0 * SZ_PACKX1;
    _Float16* pWxB  = PKX + 1 * SZ_PACKX1;

    k_wcombo<<<(GD_ * 16 + 255) / 256, 256, 0, stream>>>(Wih_m, Wedge, WC, DMAX);
    k_dmax  <<<(NB_ * LSEQ_ + 255) / 256, 256, 0, stream>>>(walks, ndeg, DMAX);
    k_pack  <<<(5 * GD_ * OUTF_ + 255) / 256, 256, 0, stream>>>(Whh_wf, Whh_wb, Whh_m, Wih_m, PK);
    k_packx <<<(12 * 512 + 255) / 256, 256, 0, stream>>>(Wih_wf, bih_wf, bhh_wf, pWxF);
    k_packx <<<(12 * 512 + 255) / 256, 256, 0, stream>>>(Wih_wb, bih_wb, bhh_wb, pWxB);
    k_xge   <<<(N_EDGES_ * GD_ + 255) / 256, 256, 0, stream>>>(e, WC, bih_m, bhh_m, XGE);
    k_xgh   <<<(N_NODES_ * GD_ + 255) / 256, 256, 0, stream>>>(h, Wih_m, bih_m, bhh_m, ndeg, DMAX, XGH);

    k_rum<<<NTILE_, 32, 0, stream>>>(
        walks, eids,
        bhh_wf, bhh_wb, bhh_m,
        XGH, XGE,
        pWhhF, pWhhB, pWhhM, pW1, pW2,
        pWxF, pWxB,
        (float*)d_out);
}
